// PathModel_11536282157158
// MI455X (gfx1250) — compile-verified, matrix operand order checked by reading, not run on hardware
//
#include <hip/hip_runtime.h>

// ---------------- problem constants ----------------
#define B_    32
#define N_    4096
#define P_    256
#define L_    16
#define DR_   128
#define IN_   256          // 2*DR
#define H_    128
#define G_    512          // 4*H
#define NPATH (B_*P_)      // 8192
#define NT_   2            // WMMA N-tiles per wave (16 waves x 32 cols)
#define NTHR  512          // threads per LSTM workgroup (16 waves)

// ---------------- vector types ----------------
typedef __attribute__((ext_vector_type(16))) __bf16         v16bf;
typedef __attribute__((ext_vector_type(8)))  float          v8f;
typedef __attribute__((ext_vector_type(4)))  unsigned int   u32x4;
typedef __attribute__((ext_vector_type(8)))  int            i32x8;
typedef __attribute__((ext_vector_type(4)))  int            i32x4;
typedef __attribute__((ext_vector_type(4)))  float          f32x4;
typedef __attribute__((ext_vector_type(4)))  unsigned short u16x4;

union FragU {            // 32 bytes: one bf16 A/B fragment (16 elems / lane)
    u32x4 q[2];
    v16bf b;
};

static __device__ __forceinline__ void load_frag(FragU& f, const unsigned short* p) {
    f.q[0] = *(const u32x4*)(p);
    f.q[1] = *(const u32x4*)(p + 16);
}

// float -> bf16 (round to nearest even), stored as ushort
static __device__ __forceinline__ unsigned short f2bf(float f) {
    union { float f; unsigned int u; } c; c.f = f;
    unsigned int r = (c.u + 0x7FFFu + ((c.u >> 16) & 1u)) >> 16;
    return (unsigned short)r;
}

static __device__ __forceinline__ float sigmoidf_(float x) {
    return 1.0f / (1.0f + __expf(-x));
}

// ---------------- workspace layout (bytes) ----------------
#define WS_WIH   0                               // bf16 [512][256]  = 262144
#define WS_WHH   (262144)                        // bf16 [512][128]  = 131072
#define WS_BIAS  (262144 + 131072)               // f32  [512]       = 2048
#define WS_LAST  (262144 + 131072 + 2048)        // f32  [8192][128] = 4 MiB

// ---------------- LDS layout (bytes, dynamic) ----------------
#define SM_X     0                               // ushort [16][256]  = 8192
#define SM_H     (8192)                          // ushort [16][128]  = 4096
#define SM_C     (8192 + 4096)                   // float  [16][128]  = 8192
#define SM_G     (8192 + 4096 + 8192)            // float  [16][512]  = 32768
#define SM_LEN   (SM_G + 32768)                  // int    [16]       = 64
#define SM_BIAS  (SM_LEN + 64)                   // float  [512]      = 2048
#define SM_TOTAL (SM_BIAS + 2048)                // 55360 B -> 2 WGs / WGP

// TDM availability (arity differs between clang-22/ROCm7.2 and clang-23)
#if defined(__has_builtin)
#  if __has_builtin(__builtin_amdgcn_tensor_load_to_lds) && \
      __has_builtin(__builtin_amdgcn_s_wait_tensorcnt)
#    define PM_HAVE_TDM 1
#  endif
#endif
#ifndef PM_HAVE_TDM
#  define PM_HAVE_TDM 0
#endif

// =====================================================================
// Kernel 1: convert weights to bf16, combine biases
// =====================================================================
__global__ void pm_prep_kernel(const float* __restrict__ W_ih,
                               const float* __restrict__ W_hh,
                               const float* __restrict__ b_ih,
                               const float* __restrict__ b_hh,
                               unsigned short* __restrict__ wih_bf,
                               unsigned short* __restrict__ whh_bf,
                               float* __restrict__ bias) {
    int g = blockIdx.x;
    int t = threadIdx.x;
    wih_bf[g * IN_ + t] = f2bf(W_ih[g * IN_ + t]);
    if (t < H_) whh_bf[g * H_ + t] = f2bf(W_hh[g * H_ + t]);
    if (t == 0) bias[g] = b_ih[g] + b_hh[g];
}

// =====================================================================
// Kernel 2: fused gather + bf16 WMMA LSTM over 16 steps
// grid = 512 workgroups (16 paths each) x 512 threads (16 waves)
// wave w owns gate columns [w*32, w*32+32) -> 2 WMMA N-tiles
// ALL weight B-fragments are VGPR-resident (W_ih 128 + W_hh 64 VGPRs/lane);
// the inner loop only streams A-fragments from LDS.
// =====================================================================
__global__ void __launch_bounds__(NTHR, 1)
pm_lstm_kernel(const float* __restrict__ graph_embed,
               const float* __restrict__ rel_embeds,
               const int*   __restrict__ node_idx,
               const int*   __restrict__ rel_idx,
               const int*   __restrict__ path_lens,
               const unsigned short* __restrict__ wih_bf,
               const unsigned short* __restrict__ whh_bf,
               const float* __restrict__ bias,
               float* __restrict__ last) {
    extern __shared__ char smem[];
    unsigned short* sX = (unsigned short*)(smem + SM_X);   // x tile   [16 paths][256]
    unsigned short* sH = (unsigned short*)(smem + SM_H);   // h state  [16][128] bf16
    float*          sC = (float*)(smem + SM_C);            // c state  [16][128]
    float*          sG = (float*)(smem + SM_G);            // gates    [16][512]
    int*          sLen = (int*)(smem + SM_LEN);            // path lens[16]
    float*       sBias = (float*)(smem + SM_BIAS);         // combined bias [512]

    const int tid      = threadIdx.x;
    const int wave     = tid >> 5;
    const int lane     = tid & 31;
    const int lrow     = lane & 15;    // fragment row (A) / col (B)
    const int lhalf    = lane >> 4;    // fragment half selects K runs
    const int pathbase = blockIdx.x * 16;

    // ---- stage combined bias (2 KB) into LDS via Tensor Data Mover ----
#if PM_HAVE_TDM
    if (wave == 0) {
        // D# group0: count=1 | lds_addr | 57-bit global addr | type=2
        unsigned long long ga = (unsigned long long)(const void*)bias;
        u32x4 g0;
        g0[0] = 1u;                                        // count=1 (valid), user mode
        g0[1] = (unsigned)SM_BIAS;                         // LDS byte address
        g0[2] = (unsigned)(ga & 0xffffffffull);            // global addr [31:0]
        g0[3] = (unsigned)((ga >> 32) & 0x01ffffffull) | 0x80000000u;  // addr[56:32] | type=2
        // D# group1: 1-D tile of 256 x 8-byte elements (2 KB)
        i32x8 g1;
        g1[0] = 0x00030000;        // workgroup_mask=0, data_size=3 (8B), no flags
        g1[1] = 0x01000000;        // abar_addr=0, tensor_dim0[15:0]=256
        g1[2] = 0x00010000;        // tensor_dim0 hi=0, tensor_dim1=1
        g1[3] = 0x01000000;        // tensor_dim1 hi=0, tile_dim0=256
        g1[4] = 0;                 // tile_dim1=0 (unused), tile_dim2=0
        g1[5] = 256;               // tensor_dim0_stride lo
        g1[6] = 0;
        g1[7] = 0;
        i32x4 gz = {0, 0, 0, 0};
#if __clang_major__ >= 23
        i32x8 gz8 = {0, 0, 0, 0, 0, 0, 0, 0};
        __builtin_amdgcn_tensor_load_to_lds(g0, g1, gz, gz, gz8, 0);
#else
        __builtin_amdgcn_tensor_load_to_lds(g0, g1, gz, gz, 0);
#endif
    }
#else
    if (tid < G_) sBias[tid] = bias[tid];
#endif

    // ---- weight B-fragments resident in VGPRs (overlaps TDM DMA) ----
    __builtin_prefetch(wih_bf + (wave * 32 + lrow) * IN_, 0, 0);
    FragU wihf[NT_][8];   // W_ih: K = 256 -> 8 ktiles
    FragU whhf[NT_][4];   // W_hh: K = 128 -> 4 ktiles
    #pragma unroll
    for (int nt = 0; nt < NT_; ++nt) {
        const int col = wave * 32 + nt * 16 + lrow;   // gate row
        #pragma unroll
        for (int kt = 0; kt < 8; ++kt)
            load_frag(wihf[nt][kt], wih_bf + col * IN_ + kt * 32 + lhalf * 8);
        #pragma unroll
        for (int kt = 0; kt < 4; ++kt)
            load_frag(whhf[nt][kt], whh_bf + col * H_ + kt * 32 + lhalf * 8);
    }

    // ---- init h, c, path lens ----
    for (int i = tid; i < 16 * H_; i += NTHR) { sC[i] = 0.0f; sH[i] = 0; }
    if (tid < 16) sLen[tid] = path_lens[pathbase + tid];

#if PM_HAVE_TDM
    __builtin_amdgcn_s_wait_tensorcnt(0);  // wave 0 drains its TDM op
#endif
    __syncthreads();                       // publishes sBias/sC/sH to all waves

    // per-lane A-fragment base pointers (step-invariant)
    const unsigned short* xbase = sX + lrow * IN_ + lhalf * 8;
    const unsigned short* hbase = sH + lrow * H_ + lhalf * 8;

    // gather assignment: 32 threads per path row, 8 cols each
    const int gm    = tid >> 5;       // path row 0..15
    const int gseg  = tid & 31;       // 8-col segment
    const int gpath = pathbase + gm;
    const int gb    = gpath >> 8;     // batch  (P_ == 256)

    for (int s = 0; s < L_; ++s) {
        // ================= gather x[s] -> LDS (bf16, masked) =================
        {
            const int plen = sLen[gm];
            const float* src;
            bool v;
            if (gseg < 16) {  // node half: cols [0,128)
                const int ni = node_idx[gpath * L_ + s];
                src = graph_embed + ((size_t)gb * N_ + ni) * DR_ + (gseg << 3);
                v = (s < plen);
            } else {          // rel half: cols [128,256)
                const int ri = rel_idx[gpath * L_ + s];
                src = rel_embeds + (size_t)ri * DR_ + ((gseg - 16) << 3);
                v = (s + 1 < plen);
            }
            const f32x4* s4 = (const f32x4*)src;
            #pragma unroll
            for (int u4 = 0; u4 < 2; ++u4) {
                f32x4 vv = {0.0f, 0.0f, 0.0f, 0.0f};
                if (v) vv = s4[u4];
                u16x4 o;
                o[0] = f2bf(vv[0]); o[1] = f2bf(vv[1]);
                o[2] = f2bf(vv[2]); o[3] = f2bf(vv[3]);
                *(u16x4*)(sX + (gm << 8) + (gseg << 3) + u4 * 4) = o;
            }
        }
        __syncthreads();

        // ================= gates = x*W_ih^T + h*W_hh^T (WMMA, zero C) ========
        v8f acc[NT_];
        #pragma unroll
        for (int nt = 0; nt < NT_; ++nt) {
            #pragma unroll
            for (int r = 0; r < 8; ++r) acc[nt][r] = 0.0f;
        }

        // A-fragment double buffer shared by both phases (only LDS loads in loop)
        FragU ab[2];
        load_frag(ab[0], xbase);
        // x contribution: K = 256 -> 8 ktiles
        #pragma unroll
        for (int kt = 0; kt < 8; ++kt) {
            const int cur = kt & 1, nxt = cur ^ 1;
            if (kt < 7) load_frag(ab[nxt], xbase + (kt + 1) * 32);
            else        load_frag(ab[nxt], hbase);          // prefetch h kt=0
            #pragma unroll
            for (int nt = 0; nt < NT_; ++nt) {
                acc[nt] = __builtin_amdgcn_wmma_f32_16x16x32_bf16(
                    false, ab[cur].b, false, wihf[nt][kt].b, (short)0, acc[nt],
                    false, false);
            }
        }
        // h contribution: K = 128 -> 4 ktiles
        #pragma unroll
        for (int kt = 0; kt < 4; ++kt) {
            const int cur = kt & 1, nxt = cur ^ 1;   // continues parity from above
            if (kt < 3) load_frag(ab[nxt], hbase + (kt + 1) * 32);
            #pragma unroll
            for (int nt = 0; nt < NT_; ++nt) {
                acc[nt] = __builtin_amdgcn_wmma_f32_16x16x32_bf16(
                    false, ab[cur].b, false, whhf[nt][kt].b, (short)0, acc[nt],
                    false, false);
            }
        }

        // scatter gate tile to LDS: acc[nt][r] -> (m = r+8*half, col)
        #pragma unroll
        for (int nt = 0; nt < NT_; ++nt) {
            const int col = wave * 32 + nt * 16 + lrow;
            #pragma unroll
            for (int r = 0; r < 8; ++r)
                sG[(r + 8 * lhalf) * G_ + col] = acc[nt][r];
        }
        __syncthreads();

        // ================= elementwise LSTM cell update (bias added here) ====
        for (int e = tid; e < 16 * H_; e += NTHR) {
            const int m = e >> 7;          // path row
            const int j = e & 127;         // hidden unit
            const float ig = sG[m * G_ + j]            + sBias[j];
            const float fg = sG[m * G_ + H_ + j]       + sBias[H_ + j];
            const float gg = sG[m * G_ + 2 * H_ + j]   + sBias[2 * H_ + j];
            const float og = sG[m * G_ + 3 * H_ + j]   + sBias[3 * H_ + j];
            const float c  = sigmoidf_(fg) * sC[e] + sigmoidf_(ig) * tanhf(gg);
            sC[e] = c;
            const float h = sigmoidf_(og) * tanhf(c);
            sH[e] = f2bf(h);
            if (s == sLen[m] - 1)          // last valid output of this path
                last[(size_t)(pathbase + m) * H_ + j] = h;
        }
        __syncthreads();
    }
}

// =====================================================================
// Kernel 3: fused attention + out_proj (tiny, f32 VALU)
// grid = 32 blocks (one per batch) x 128 threads
// =====================================================================
__global__ void pm_attn_kernel(const float* __restrict__ node_opt,
                               const float* __restrict__ in_proj_w,   // [384][128]
                               const float* __restrict__ in_proj_b,   // [384]
                               const float* __restrict__ out_proj_w,  // [128][128]
                               const float* __restrict__ out_proj_b,  // [128]
                               const float* __restrict__ last,        // [8192][128]
                               float* __restrict__ out) {             // [32][128]
    __shared__ float sq[128];
    __shared__ float sqk[128];
    __shared__ float sattn[256];
    __shared__ float sctxp[128];
    __shared__ float sctx[128];
    __shared__ float sred[128];

    const int b = blockIdx.x;
    const int t = threadIdx.x;

    // q_j = Wq[j,:] . node_opt[b] + bq[j]
    float q = in_proj_b[t];
    const float* no = node_opt + b * H_;
    for (int d = 0; d < H_; ++d) q += in_proj_w[t * H_ + d] * no[d];
    sq[t] = q;
    __syncthreads();

    // qk_d = sum_j q_j * Wk[j,d];   c0 = sum_j q_j * bk_j
    float qk = 0.0f;
    for (int j = 0; j < H_; ++j) qk += sq[j] * in_proj_w[(H_ + j) * H_ + t];
    sqk[t] = qk;
    sred[t] = sq[t] * in_proj_b[H_ + t];
    __syncthreads();
    for (int off = 64; off > 0; off >>= 1) {
        if (t < off) sred[t] += sred[t + off];
        __syncthreads();
    }
    const float c0 = sred[0];
    __syncthreads();

    // logits over P = 256 (2 per thread)
    const float scale = 0.08838834764831845f;   // 1/sqrt(128)
    float lg[2];
    #pragma unroll
    for (int i = 0; i < 2; ++i) {
        const int p = t + i * 128;
        const float* lp = last + ((size_t)b * P_ + p) * H_;
        float l = c0;
        for (int d = 0; d < H_; ++d) l += sqk[d] * lp[d];
        lg[i] = l * scale;
    }
    sred[t] = fmaxf(lg[0], lg[1]);
    __syncthreads();
    for (int off = 64; off > 0; off >>= 1) {
        if (t < off) sred[t] = fmaxf(sred[t], sred[t + off]);
        __syncthreads();
    }
    const float mx = sred[0];
    __syncthreads();
    const float e0 = __expf(lg[0] - mx);
    const float e1 = __expf(lg[1] - mx);
    sattn[t] = e0; sattn[t + 128] = e1;
    sred[t] = e0 + e1;
    __syncthreads();
    for (int off = 64; off > 0; off >>= 1) {
        if (t < off) sred[t] += sred[t + off];
        __syncthreads();
    }
    const float inv = 1.0f / sred[0];
    __syncthreads();

    // ctxpre_d = sum_p softmax_p * last[b,p,d]
    float cp = 0.0f;
    for (int p = 0; p < P_; ++p) cp += sattn[p] * last[((size_t)b * P_ + p) * H_ + t];
    sctxp[t] = cp * inv;
    __syncthreads();

    // ctx_j = Wv[j,:] . ctxpre + bv[j]
    float cx = in_proj_b[2 * H_ + t];
    for (int d = 0; d < H_; ++d) cx += in_proj_w[(2 * H_ + t) * H_ + d] * sctxp[d];
    sctx[t] = cx;
    __syncthreads();

    // out_i = out_proj_w[i,:] . ctx + out_proj_b[i]
    float o = out_proj_b[t];
    for (int j = 0; j < H_; ++j) o += out_proj_w[t * H_ + j] * sctx[j];
    out[(size_t)b * H_ + t] = o;
}

// =====================================================================
extern "C" void kernel_launch(void* const* d_in, const int* in_sizes, int n_in,
                              void* d_out, int out_size, void* d_ws, size_t ws_size,
                              hipStream_t stream) {
    const float* graph_embed = (const float*)d_in[0];
    const float* node_opt    = (const float*)d_in[1];
    const float* rel_embeds  = (const float*)d_in[2];
    const float* W_ih        = (const float*)d_in[3];
    const float* W_hh        = (const float*)d_in[4];
    const float* b_ih        = (const float*)d_in[5];
    const float* b_hh        = (const float*)d_in[6];
    const float* in_proj_w   = (const float*)d_in[7];
    const float* in_proj_b   = (const float*)d_in[8];
    const float* out_proj_w  = (const float*)d_in[9];
    const float* out_proj_b  = (const float*)d_in[10];
    const int*   node_idx    = (const int*)d_in[11];
    const int*   rel_idx     = (const int*)d_in[12];
    const int*   path_lens   = (const int*)d_in[13];

    char* ws = (char*)d_ws;
    unsigned short* wih_bf = (unsigned short*)(ws + WS_WIH);
    unsigned short* whh_bf = (unsigned short*)(ws + WS_WHH);
    float*          bias   = (float*)(ws + WS_BIAS);
    float*          last   = (float*)(ws + WS_LAST);
    float*          out    = (float*)d_out;

    (void)hipFuncSetAttribute((const void*)pm_lstm_kernel,
                              hipFuncAttributeMaxDynamicSharedMemorySize,
                              (int)SM_TOTAL);

    pm_prep_kernel<<<G_, IN_, 0, stream>>>(W_ih, W_hh, b_ih, b_hh,
                                           wih_bf, whh_bf, bias);

    pm_lstm_kernel<<<NPATH / 16, NTHR, SM_TOTAL, stream>>>(
        graph_embed, rel_embeds, node_idx, rel_idx, path_lens,
        wih_bf, whh_bf, bias, last);

    pm_attn_kernel<<<B_, H_, 0, stream>>>(node_opt, in_proj_w, in_proj_b,
                                          out_proj_w, out_proj_b, last, out);
}
